// MultiHeadSelfAttention_6889127542899
// MI455X (gfx1250) — compile-verified, matrix operand order checked by reading, not run on hardware
//
#include <hip/hip_runtime.h>
#include <stdint.h>

// ---------------------------------------------------------------------------
// MI455X (gfx1250) multi-head self-attention, f16 WMMA pipeline with
// TDM (tensor_load_to_lds) double-buffered GEMM staging.
// B=8, S=1024, D=768, H=12, DH=64.
// ---------------------------------------------------------------------------

typedef __attribute__((ext_vector_type(16))) _Float16 v16h;
typedef __attribute__((ext_vector_type(8)))  _Float16 h8;
typedef __attribute__((ext_vector_type(4)))  _Float16 h4;
typedef __attribute__((ext_vector_type(8)))  float    v8f;
typedef __attribute__((ext_vector_type(4)))  float    f4;

#define WMMA16(a, b, c) \
  __builtin_amdgcn_wmma_f32_16x16x32_f16(false, (a), false, (b), (short)0, (c), false, false)

union Frag16 { v16h v; h8 h[2]; };

// Fragment loader, per ISA 7.12.2: lane holds K=koff..koff+7 and K=koff+16..koff+23,
// koff = (lane>>4)*8.  Works for global or LDS pointers (two 16-byte loads).
__device__ inline v16h load_frag(const _Float16* __restrict__ base, int stride, int lane) {
  const _Float16* p = base + (lane & 15) * stride + ((lane >> 4) << 3);
  Frag16 u;
  u.h[0] = *(const h8*)(p);
  u.h[1] = *(const h8*)(p + 16);
  return u.v;
}

__device__ inline v16h lds_frag(const _Float16* p) {   // p already lane-resolved
  Frag16 u;
  u.h[0] = *(const h8*)(p);
  u.h[1] = *(const h8*)(p + 16);
  return u.v;
}

// ---------------------------------------------------------------------------
// TDM tile staging: 128 rows x 32 f16 from a row-major matrix (row stride 768)
// into LDS with 16B padding after each 64B row (LDS row stride 40 f16 = 80B).
// ---------------------------------------------------------------------------
#define TSTRIDE 40   // f16 elements per LDS tile row (32 data + 8 pad)

#if __has_builtin(__builtin_amdgcn_tensor_load_to_lds)
#define USE_TDM 1
typedef __attribute__((ext_vector_type(4))) unsigned int u32x4;
typedef __attribute__((ext_vector_type(8))) int i32x8;
typedef __attribute__((ext_vector_type(4))) int i32x4;

__device__ inline void tdm_tile_load(const _Float16* g, unsigned int lds_addr) {
  const unsigned long long ga = (unsigned long long)(uintptr_t)g;
  u32x4 g0;
  g0[0] = 1u;                                   // count=1, user mode, no gather
  g0[1] = lds_addr;                             // lds_addr (bytes)
  g0[2] = (unsigned int)ga;                     // global_addr[31:0]
  g0[3] = (unsigned int)((ga >> 32) & 0x01FFFFFFu) | 0x80000000u; // ga[56:32] | type=2
  i32x8 g1;
  g1[0] = 0x06D10000;  // wg_mask=0 | data_size=2B | pad_en | interval=16dw | amount=4dw
  g1[1] = 32 << 16;    // tensor_dim0 = 32 (tile-local view: no OOB)
  g1[2] = 128 << 16;   // tensor_dim1 = 128
  g1[3] = 32 << 16;    // tile_dim0 = 32
  g1[4] = 128;         // tile_dim1 = 128, tile_dim2 = 0
  g1[5] = 768;         // tensor_dim0_stride = 768 elements
  g1[6] = 0;
  g1[7] = 0;
  i32x4 z = {0, 0, 0, 0};
#if __clang_major__ >= 23
  i32x8 z8 = {0, 0, 0, 0, 0, 0, 0, 0};
  __builtin_amdgcn_tensor_load_to_lds(g0, g1, z, z, z8, 0);
#else
  __builtin_amdgcn_tensor_load_to_lds(g0, g1, z, z, 0);
#endif
}
#else
#define USE_TDM 0
#endif

__device__ inline void wait_tensor_all() {
#if __has_builtin(__builtin_amdgcn_s_wait_tensorcnt)
  __builtin_amdgcn_s_wait_tensorcnt(0);
#else
  asm volatile("s_wait_tensorcnt 0x0" ::: "memory");
#endif
}

// Fallback staging: identical padded LDS layout, 256 threads, 16B each x2.
__device__ inline void sync_tile_copy(const _Float16* __restrict__ g,
                                      _Float16* __restrict__ l, int t) {
  const int row = t >> 1, half = t & 1;
  *(h8*)(l + row * TSTRIDE + half * 16) = *(const h8*)(g + row * 768 + half * 16);
}

// ---------------------------------------------------------------------------
// f32 -> f16 conversion (vectorized x4)
// ---------------------------------------------------------------------------
__global__ __launch_bounds__(256)
void cvt_f32_f16(const float* __restrict__ in, _Float16* __restrict__ out) {
  size_t i = (size_t)blockIdx.x * 256 + threadIdx.x;
  f4 x = *(const f4*)(in + i * 4);
  h4 y;
  y[0] = (_Float16)x[0]; y[1] = (_Float16)x[1];
  y[2] = (_Float16)x[2]; y[3] = (_Float16)x[3];
  *(h4*)(out + i * 4) = y;
}

// ---------------------------------------------------------------------------
// W (768x768 f32, row-major) -> W^T (768x768 f16, row-major)
// ---------------------------------------------------------------------------
__global__ __launch_bounds__(256)
void transpose_w(const float* __restrict__ W, _Float16* __restrict__ Wt) {
  int idx = blockIdx.x * 256 + threadIdx.x;   // 0 .. 589823
  int n = idx / 768;
  int k = idx % 768;
  Wt[idx] = (_Float16)W[k * 768 + n];
}

// ---------------------------------------------------------------------------
// GEMM: Y = A(8192x768 f16) @ Bt^T + bias, TDM-staged, LDS double-buffered.
// 8 waves (4x2), block tile 128x128, per-wave 32x64, K-step 32.
// mode 0: f16 out [B][H][S][DH] (Q,K)   mode 1: f16 out [B][H][DH][S] (V^T)
// mode 2: f32 out [8192][768] (final projection)
// ---------------------------------------------------------------------------
__global__ __launch_bounds__(256)
void gemm_wmma(const _Float16* __restrict__ A, const _Float16* __restrict__ Bt,
               const float* __restrict__ bias, void* __restrict__ out, int mode) {
  __shared__ _Float16 atile[2][128 * TSTRIDE];   // 2 x 10 KB
  __shared__ _Float16 btile[2][128 * TSTRIDE];   // 2 x 10 KB

  const int lane = threadIdx.x & 31;
  const int wid  = threadIdx.x >> 5;
  const int wm   = wid & 3;          // 4 waves over M
  const int wn   = wid >> 2;         // 2 waves over N
  const int m0b  = blockIdx.x * 128;
  const int n0b  = blockIdx.y * 128;
  const int koff = (lane >> 4) << 3;

  v8f acc[2][4] = {};

  // stage K-step 0
#if USE_TDM
  if (wid == 0) {
    tdm_tile_load(A  + (size_t)m0b * 768, (unsigned int)(uintptr_t)&atile[0][0]);
    tdm_tile_load(Bt + (size_t)n0b * 768, (unsigned int)(uintptr_t)&btile[0][0]);
  }
#else
  sync_tile_copy(A  + (size_t)m0b * 768, &atile[0][0], threadIdx.x);
  sync_tile_copy(Bt + (size_t)n0b * 768, &btile[0][0], threadIdx.x);
#endif

  for (int ki = 0; ki < 24; ++ki) {
    const int cur = ki & 1;
#if USE_TDM
    if (wid == 0) wait_tensor_all();     // tile ki landed (issuing wave's TENSORcnt)
#endif
    __syncthreads();                     // publish tile ki; all waves done with buf cur^1

    if (ki < 23) {                       // prefetch tile ki+1 into the other buffer
      const int kk = (ki + 1) * 32;
#if USE_TDM
      if (wid == 0) {
        tdm_tile_load(A  + (size_t)m0b * 768 + kk, (unsigned int)(uintptr_t)&atile[cur ^ 1][0]);
        tdm_tile_load(Bt + (size_t)n0b * 768 + kk, (unsigned int)(uintptr_t)&btile[cur ^ 1][0]);
      }
#else
      sync_tile_copy(A  + (size_t)m0b * 768 + kk, &atile[cur ^ 1][0], threadIdx.x);
      sync_tile_copy(Bt + (size_t)n0b * 768 + kk, &btile[cur ^ 1][0], threadIdx.x);
#endif
    }

    // compute from buffer `cur` (ds_load_b128 fragments)
    const _Float16* ap = &atile[cur][0];
    const _Float16* bp = &btile[cur][0];
    v16h a0 = lds_frag(ap + (wm * 32 +      (lane & 15)) * TSTRIDE + koff);
    v16h a1 = lds_frag(ap + (wm * 32 + 16 + (lane & 15)) * TSTRIDE + koff);
#pragma unroll
    for (int nf = 0; nf < 4; ++nf) {
      v16h b = lds_frag(bp + (wn * 64 + nf * 16 + (lane & 15)) * TSTRIDE + koff);
      acc[0][nf] = WMMA16(a0, b, acc[0][nf]);
      acc[1][nf] = WMMA16(a1, b, acc[1][nf]);
    }
  }

  // epilogue
  const int nlo   = lane & 15;
  const int mbase = (lane >> 4) << 3;
#pragma unroll
  for (int mf = 0; mf < 2; ++mf) {
#pragma unroll
    for (int nf = 0; nf < 4; ++nf) {
      const int gn = n0b + wn * 64 + nf * 16 + nlo;
      const float bv = bias[gn];
#pragma unroll
      for (int v = 0; v < 8; ++v) {
        const int gm = m0b + wm * 32 + mf * 16 + mbase + v;
        const float val = acc[mf][nf][v] + bv;
        if (mode == 2) {
          ((float*)out)[(size_t)gm * 768 + gn] = val;
        } else {
          const int b  = gm >> 10, s  = gm & 1023;
          const int h  = gn >> 6,  dh = gn & 63;
          const size_t bh = (size_t)b * 12 + h;
          const size_t off = (mode == 0) ? (bh * 1024 + s) * 64 + dh
                                         : (bh * 64 + dh) * 1024 + s;
          ((_Float16*)out)[off] = (_Float16)val;
        }
      }
    }
  }
}

// ---------------------------------------------------------------------------
// Attention: one wave per (b,h,16-query tile).  grid = 96*64 = 6144.
// Scores (16x1024, scaled) -> LDS f16; in-place softmax; P @ V^T via WMMA.
// ---------------------------------------------------------------------------
__global__ __launch_bounds__(32)
void attn_wmma(const _Float16* __restrict__ Q, const _Float16* __restrict__ K,
               const _Float16* __restrict__ Vt, _Float16* __restrict__ Ctx) {
  __shared__ _Float16 sc[16 * 1024];   // 32 KB: scores, then P in place

  const int lane = threadIdx.x;
  const int qt = blockIdx.x & 63;
  const int bh = blockIdx.x >> 6;          // 0..95
  const int b  = bh / 12, h = bh % 12;

  const _Float16* Qb = Q  + ((size_t)bh * 1024 + qt * 16) * 64;
  const _Float16* Kb = K  + (size_t)bh * 1024 * 64;
  const _Float16* Vb = Vt + (size_t)bh * 64 * 1024;

  const int nlo   = lane & 15;
  const int koff  = (lane >> 4) << 3;
  const int mbase = (lane >> 4) << 3;

  // Q tile fragments (16 x 64 = two 16x32 A-frags), held in registers
  const v16h qf0 = load_frag(Qb + 0,  64, lane);
  const v16h qf1 = load_frag(Qb + 32, 64, lane);

  // ---- pass 1: scores = (Q K^T) / sqrt(64) -> LDS ----
  for (int jt = 0; jt < 64; ++jt) {
    if (jt + 1 < 64)  // global_prefetch_b8 for next key tile
      __builtin_prefetch(Kb + (size_t)((jt + 1) * 16 + nlo) * 64, 0, 1);
    const _Float16* Krow = Kb + (size_t)(jt * 16) * 64;
    v16h k0 = load_frag(Krow + 0,  64, lane);
    v16h k1 = load_frag(Krow + 32, 64, lane);
    v8f s = {};
    s = WMMA16(qf0, k0, s);
    s = WMMA16(qf1, k1, s);
#pragma unroll
    for (int v = 0; v < 8; ++v)
      sc[(mbase + v) * 1024 + jt * 16 + nlo] = (_Float16)(s[v] * 0.125f);
  }
  __syncthreads();

  // ---- pass 2: in-place softmax.  lane handles row (lane&15), half (lane>>4) ----
  const int r  = lane & 15;
  const int c0 = (lane >> 4) * 512;
  float mx = -1e30f;
  for (int c = 0; c < 512; c += 8) {
    h8 x = *(const h8*)&sc[r * 1024 + c0 + c];
#pragma unroll
    for (int i = 0; i < 8; ++i) mx = fmaxf(mx, (float)x[i]);
  }
  mx = fmaxf(mx, __shfl_xor(mx, 16, 32));   // combine the two column halves

  float sum = 0.f;
  for (int c = 0; c < 512; c += 8) {
    h8 x = *(const h8*)&sc[r * 1024 + c0 + c];
    h8 y;
#pragma unroll
    for (int i = 0; i < 8; ++i) {
      float e = __expf((float)x[i] - mx);
      sum += e;
      y[i] = (_Float16)e;
    }
    *(h8*)&sc[r * 1024 + c0 + c] = y;       // same lane, same address: safe
  }
  sum += __shfl_xor(sum, 16, 32);
  const float inv = 1.0f / sum;
  __syncthreads();

  // ---- pass 3: out = P @ V  (A-frags from LDS, B-frags from V^T global) ----
  v8f acc[4] = {};
  for (int kt = 0; kt < 32; ++kt) {
    Frag16 pa;
    const _Float16* pp = &sc[(lane & 15) * 1024 + kt * 32 + koff];
    pa.h[0] = *(const h8*)pp;
    pa.h[1] = *(const h8*)(pp + 16);
#pragma unroll
    for (int nf = 0; nf < 4; ++nf) {
      v16h vf = load_frag(Vb + (size_t)(nf * 16) * 1024 + kt * 32, 1024, lane);
      acc[nf] = WMMA16(pa.v, vf, acc[nf]);
    }
  }

  // epilogue: normalize rows, write ctx[b][s][h*64+dh] (f16)
#pragma unroll
  for (int v = 0; v < 8; ++v) {
    const float rinv = __shfl(inv, mbase + v, 32);  // row sum lives in lane (row&15)
    const int sidx = qt * 16 + mbase + v;
#pragma unroll
    for (int nf = 0; nf < 4; ++nf) {
      const int dh = nf * 16 + nlo;
      Ctx[((size_t)(b * 1024 + sidx)) * 768 + h * 64 + dh] = (_Float16)(acc[nf][v] * rinv);
    }
  }
}

// ---------------------------------------------------------------------------
// Host launcher
// ---------------------------------------------------------------------------
extern "C" void kernel_launch(void* const* d_in, const int* in_sizes, int n_in,
                              void* d_out, int out_size, void* d_ws, size_t ws_size,
                              hipStream_t stream) {
  (void)in_sizes; (void)n_in; (void)out_size; (void)ws_size;

  const float* x  = (const float*)d_in[0];
  const float* Wq = (const float*)d_in[1];
  const float* bq = (const float*)d_in[2];
  const float* Wk = (const float*)d_in[3];
  const float* bk = (const float*)d_in[4];
  const float* Wv = (const float*)d_in[5];
  const float* bv = (const float*)d_in[6];
  const float* Wo = (const float*)d_in[7];
  const float* bo = (const float*)d_in[8];
  float* out = (float*)d_out;

  // workspace layout (f16 elements)
  const size_t XN = 8192ull * 768;     // 6291456
  const size_t WN = 768ull * 768;      // 589824
  _Float16* Xh  = (_Float16*)d_ws;     // also reused as Ctx after QKV
  _Float16* Wqt = Xh  + XN;
  _Float16* Wkt = Wqt + WN;
  _Float16* Wvt = Wkt + WN;
  _Float16* Wot = Wvt + WN;
  _Float16* Qh  = Wot + WN;
  _Float16* Kh  = Qh  + XN;
  _Float16* Vth = Kh  + XN;
  _Float16* Ctx = Xh;                  // X dead after QKV projections

  cvt_f32_f16<<<6144, 256, 0, stream>>>(x, Xh);
  transpose_w<<<2304, 256, 0, stream>>>(Wq, Wqt);
  transpose_w<<<2304, 256, 0, stream>>>(Wk, Wkt);
  transpose_w<<<2304, 256, 0, stream>>>(Wv, Wvt);
  transpose_w<<<2304, 256, 0, stream>>>(Wo, Wot);

  dim3 gg(64, 6);                                        // M/128 x N/128
  gemm_wmma<<<gg, 256, 0, stream>>>(Xh, Wqt, bq, (void*)Qh,  0);
  gemm_wmma<<<gg, 256, 0, stream>>>(Xh, Wkt, bk, (void*)Kh,  0);
  gemm_wmma<<<gg, 256, 0, stream>>>(Xh, Wvt, bv, (void*)Vth, 1);

  attn_wmma<<<6144, 32, 0, stream>>>(Qh, Kh, Vth, Ctx);  // 96 (b,h) x 64 q-tiles

  gemm_wmma<<<gg, 256, 0, stream>>>(Ctx, Wot, bo, (void*)out, 2);
}
